// TemporalDependency_46213848105940
// MI455X (gfx1250) — matrix operand order repaired and checked by reading, not compiled, not run on hardware
//
#include <hip/hip_runtime.h>
#include <hip/hip_bf16.h>

// ---------------------------------------------------------------------------
// Problem constants
// ---------------------------------------------------------------------------
#define B_      16
#define T_      2048
#define RANK_   256
#define H_      1024
#define G4_     4096          // 4*H gates
#define NWG     64            // persistent grid: 64 workgroups (1 per WGP)
#define NTHR    128           // 4 waves of 32
#define KT0     40            // layer0 k-tiles: (256 + 1024)/32
#define KT1     64            // layer1 k-tiles: (1024 + 1024)/32
#define KT1L    32            // layer1 k-tiles cached in LDS (h0 half)
#define KTD     32            // decoder k-tiles: 1024/32

// dynamic LDS: layer0 full slice + layer1 first half (shorts)
#define LDS_W0_SHORTS (4 * KT0 * 512)     // 81920  (160 KB)
#define LDS_W1_SHORTS (4 * KT1L * 512)    // 65536  (128 KB)
#define LDS_BYTES     ((LDS_W0_SHORTS + LDS_W1_SHORTS) * 2)  // 294912 B

typedef __attribute__((ext_vector_type(16))) __bf16       bf16x16;
typedef __attribute__((ext_vector_type(8)))  float        floatx8;
typedef __attribute__((ext_vector_type(4)))  unsigned int u32x4v;

union ABfrag { u32x4v u[2]; bf16x16 v; };
union F8     { floatx8 v; float f[8]; };

// ---------------------------------------------------------------------------
// helpers
// ---------------------------------------------------------------------------
__device__ __forceinline__ unsigned short f2bf(float f) {
  unsigned int u = __float_as_uint(f);
  u += 0x7FFFu + ((u >> 16) & 1u);          // round-to-nearest-even
  return (unsigned short)(u >> 16);
}

__device__ __forceinline__ float sigmoidf_(float x) {
  return 1.0f / (1.0f + __expf(-x));
}

// A operand (16x32 bf16): lane m = lane&15, half = lane>>4.
// e0..15 -> K = kbase + half*16 + e
__device__ __forceinline__ bf16x16 load_a(const unsigned short* rowptr, int kbase, int half) {
  ABfrag f;
  f.u[0] = *(const u32x4v*)(rowptr + kbase + half * 16);
  f.u[1] = *(const u32x4v*)(rowptr + kbase + half * 16 + 8);
  return f.v;
}

// B operand pre-packed: contiguous 16 bf16 per lane (global or LDS pointer)
__device__ __forceinline__ bf16x16 load_b(const unsigned short* p) {
  ABfrag f;
  f.u[0] = *(const u32x4v*)(p);
  f.u[1] = *(const u32x4v*)(p + 8);
  return f.v;
}

// accumulate k-tiles [kt0, ktn) ; rowptr covers local K starting at kt0;
// bp indexed with the SAME kt (caller pre-biases the base pointer if needed)
__device__ __forceinline__ floatx8 gemm_span(floatx8 acc, const unsigned short* rowptr,
                                             const unsigned short* bp, int kt0, int ktn,
                                             int half) {
#pragma unroll 4
  for (int kt = kt0; kt < ktn; ++kt) {
    bf16x16 a = load_a(rowptr, (kt - kt0) * 32, half);
    bf16x16 b = load_b(bp + kt * 512);
    acc = __builtin_amdgcn_wmma_f32_16x16x32_bf16(false, a, false, b, (short)0, acc,
                                                  false, false);
  }
  return acc;
}

// sense-reversing grid barrier (all NWG blocks resident)
__device__ __forceinline__ void grid_barrier(unsigned int* bar, unsigned int& nbar) {
  __syncthreads();
  nbar++;
  unsigned int s = nbar & 1u;
  if (threadIdx.x == 0) {
    unsigned int prev = __hip_atomic_fetch_add(&bar[0], 1u, __ATOMIC_ACQ_REL,
                                               __HIP_MEMORY_SCOPE_AGENT);
    if (prev == NWG - 1) {
      __hip_atomic_store(&bar[0], 0u, __ATOMIC_RELAXED, __HIP_MEMORY_SCOPE_AGENT);
      __hip_atomic_store(&bar[1], s, __ATOMIC_RELEASE, __HIP_MEMORY_SCOPE_AGENT);
    } else {
      while (__hip_atomic_load(&bar[1], __ATOMIC_ACQUIRE, __HIP_MEMORY_SCOPE_AGENT) != s) {
        __builtin_amdgcn_s_sleep(2);
      }
    }
  }
  __syncthreads();
  __builtin_amdgcn_fence(__ATOMIC_ACQUIRE, "agent");
}

// ---------------------------------------------------------------------------
// setup kernels
// ---------------------------------------------------------------------------
// Pack W (rows n = output gate, cols K = [x|h]) into WMMA-B lane layout:
// dst[((nt*ktiles + kt)*32 + lane)*16 + e] = W[n = nt*16 + (lane&15)][K]
// with K = kt*32 + (lane>>4)*16 + e.
__global__ void pack_b_kernel(unsigned short* dst, const float* Wx, const float* Wh,
                              int Kx, int Kh, int ktiles, long long total) {
  long long idx = (long long)blockIdx.x * blockDim.x + threadIdx.x;
  if (idx >= total) return;
  int e    = (int)(idx & 15);
  int lane = (int)((idx >> 4) & 31);
  int kt   = (int)((idx >> 9) % ktiles);
  int nt   = (int)(idx / ((long long)ktiles * 512));
  int n    = nt * 16 + (lane & 15);
  int K    = kt * 32 + ((lane >> 4) << 4) + e;
  float v  = (K < Kx) ? Wx[(long long)n * Kx + K]
                      : Wh[(long long)n * Kh + (K - Kx)];
  dst[idx] = f2bf(v);
}

// bf16 copy of inputs Ws[:, :T-1, :] -> [b][t][k], row length RANK_
__global__ void conv_x_kernel(unsigned short* dst, const float* Ws) {
  long long i = (long long)blockIdx.x * blockDim.x + threadIdx.x;
  long long total = (long long)B_ * (T_ - 1) * RANK_;
  if (i >= total) return;
  int k = (int)(i % RANK_);
  long long bt = i / RANK_;
  int t = (int)(bt % (T_ - 1));
  int b = (int)(bt / (T_ - 1));
  dst[i] = f2bf(Ws[((long long)b * T_ + t) * RANK_ + k]);
}

__global__ void bias_kernel(float* b0, float* b1, const float* bi0, const float* bh0,
                            const float* bi1, const float* bh1) {
  int i = blockIdx.x * blockDim.x + threadIdx.x;
  if (i < G4_)            b0[i] = bi0[i] + bh0[i];
  else if (i < 2 * G4_)   b1[i - G4_] = bi1[i - G4_] + bh1[i - G4_];
}

__global__ void zero_kernel(unsigned int* p, long long nwords) {
  long long i = (long long)blockIdx.x * blockDim.x + threadIdx.x;
  if (i < nwords) p[i] = 0u;
}

// ---------------------------------------------------------------------------
// persistent fused 2-layer LSTM + decoder/loss
// ---------------------------------------------------------------------------
__global__ void __launch_bounds__(NTHR) lstm_persistent(
    const unsigned short* __restrict__ packW0,  // [256 nt][40 kt][32][16]
    const unsigned short* __restrict__ packW1,  // [256 nt][64 kt][32][16]
    const unsigned short* __restrict__ packWd,  // [16  nt][32 kt][32][16]
    const unsigned short* __restrict__ xbf,     // [16][2047][256] bf16
    const float* __restrict__ bias0, const float* __restrict__ bias1,
    float* __restrict__ c0, float* __restrict__ c1,
    unsigned short* __restrict__ h0bf,          // 2 x [16][1024] bf16
    unsigned short* __restrict__ h1bf,          // 2 x [16][1024] bf16
    float* __restrict__ decacc,                 // [16 dw][32 lane][8 r]
    unsigned int* __restrict__ bar,
    const float* __restrict__ Ws, const int* __restrict__ lengths,
    const float* __restrict__ bdec) {
  const int tid  = threadIdx.x;
  const int lane = tid & 31;
  const int q    = tid >> 5;          // wave within WG: gate index 0..3
  const int w    = blockIdx.x;        // 0..63: 16-wide slice of H
  const int half = lane >> 4;
  const int n    = lane & 15;         // D column within tile
  const int m    = n;                 // A-fragment row = batch for this lane

  __shared__ float zsh[4][8][32];     // staged gate tiles (i,f,g,o)
  extern __shared__ unsigned short smem[];
  unsigned short* ldsW0 = smem;                       // [4][KT0 ][32][16]
  unsigned short* ldsW1 = smem + LDS_W0_SHORTS;       // [4][KT1L][32][16]

  // ---- one-time cooperative weight staging: global -> LDS -----------------
  {
#pragma unroll
    for (int g = 0; g < 4; ++g) {
      const u32x4v* src = (const u32x4v*)(packW0 + (size_t)(g * 64 + w) * KT0 * 512);
      u32x4v*       dst = (u32x4v*)(ldsW0 + g * KT0 * 512);
      for (int i = tid; i < KT0 * 512 / 8; i += NTHR) dst[i] = src[i];
    }
#pragma unroll
    for (int g = 0; g < 4; ++g) {
      const u32x4v* src = (const u32x4v*)(packW1 + (size_t)(g * 64 + w) * KT1 * 512);
      u32x4v*       dst = (u32x4v*)(ldsW1 + g * KT1L * 512);
      for (int i = tid; i < KT1L * 512 / 8; i += NTHR) dst[i] = src[i];
    }
  }
  __syncthreads();

  unsigned int nbar = 0;

  const float b0v = bias0[q * H_ + w * 16 + n];
  const float b1v = bias1[q * H_ + w * 16 + n];

  // decoder wave state (WGs 0..3, dw = w*4+q in 0..15)
  const int dw = w * 4 + q;
  float acc8[8];
  int   Lr[8];
  float bdv = 0.f;
  if (w < 4) {
    bdv = bdec[dw * 16 + n];
#pragma unroll
    for (int r = 0; r < 8; ++r) {
      acc8[r] = 0.f;
      Lr[r]   = lengths[r + 8 * half] - 1;
    }
  }

  // per-wave B fragment base pointers
  const unsigned short* bp0  = ldsW0 + q * KT0 * 512 + lane * 16;           // LDS
  const unsigned short* bp1l = ldsW1 + q * KT1L * 512 + lane * 16;          // LDS
  const unsigned short* bp1g = packW1 + (size_t)(q * 64 + w) * KT1 * 512 + lane * 16;

  const int BH = B_ * H_;

  for (int t = 0; t < T_; ++t) {      // t = 0..2047 (layers run for t<2047)
    const int p = t & 1;
    unsigned short*       h0cur  = h0bf + p * BH;
    const unsigned short* h0prev = h0bf + (1 - p) * BH;
    unsigned short*       h1cur  = h1bf + p * BH;
    const unsigned short* h1prev = h1bf + (1 - p) * BH;

    // ---------------- phase 0: layer-0 step t  (+ decoder for t-1) ----------
    if (t < T_ - 1) {
      const unsigned short* xrow = xbf + ((size_t)m * (T_ - 1) + t) * RANK_;
      const unsigned short* hrow = h0prev + m * H_;
      floatx8 acc = {};
      acc = gemm_span(acc, xrow, bp0, 0, 8, half);    // K 0..255   (x) , LDS B
      acc = gemm_span(acc, hrow, bp0, 8, KT0, half);  // K 256..1279 (h), LDS B
      F8 z; z.v = acc;
#pragma unroll
      for (int r = 0; r < 8; ++r) zsh[q][r][lane] = z.f[r] + b0v;
    }
    __syncthreads();
    if (t < T_ - 1) {
      int rr = q;
#pragma unroll
      for (int s = 0; s < 2; ++s, rr += 4) {
        float iv = sigmoidf_(zsh[0][rr][lane]);
        float fv = sigmoidf_(zsh[1][rr][lane]);
        float gv = tanhf(zsh[2][rr][lane]);
        float ov = sigmoidf_(zsh[3][rr][lane]);
        int mm  = rr + 8 * half;                 // batch
        int idx = mm * H_ + w * 16 + n;          // h-column slice of this WG
        float cn = fv * c0[idx] + iv * gv;
        c0[idx] = cn;
        h0cur[idx] = f2bf(ov * tanhf(cn));
      }
    }
    // decoder + masked MSE for output index t-1 (reads h1 of step t-1)
    if (w < 4 && t > 0) {
      const unsigned short* hrow = h1prev + m * H_;
      const unsigned short* bp   = packWd + (size_t)dw * KTD * 512 + lane * 16;
      floatx8 da = {};
      da = gemm_span(da, hrow, bp, 0, KTD, half);
      F8 z; z.v = da;
#pragma unroll
      for (int r = 0; r < 8; ++r) {
        int mm   = r + 8 * half;
        int rcol = dw * 16 + n;
        float lv = z.f[r] + bdv;
        lv = lv > 0.f ? lv : 0.f;                              // relu
        float tg = Ws[((size_t)mm * T_ + t) * RANK_ + rcol];   // targets[t-1]=Ws[:,t]
        float d  = lv - tg;
        if ((t - 1) < Lr[r]) acc8[r] += d * d;
      }
    }
    grid_barrier(bar, nbar);

    // ---------------- phase 1: layer-1 step t -------------------------------
    if (t < T_ - 1) {
      const unsigned short* xrow = h0cur + m * H_;   // layer-0 output of step t
      const unsigned short* hrow = h1prev + m * H_;
      floatx8 acc = {};
      acc = gemm_span(acc, xrow, bp1l, 0, KT1L, half);    // K 0..1023 (h0_t), LDS B
      acc = gemm_span(acc, hrow, bp1g, KT1L, KT1, half);  // K 1024..2047,  global B
      F8 z; z.v = acc;
#pragma unroll
      for (int r = 0; r < 8; ++r) zsh[q][r][lane] = z.f[r] + b1v;
    }
    __syncthreads();
    if (t < T_ - 1) {
      int rr = q;
#pragma unroll
      for (int s = 0; s < 2; ++s, rr += 4) {
        float iv = sigmoidf_(zsh[0][rr][lane]);
        float fv = sigmoidf_(zsh[1][rr][lane]);
        float gv = tanhf(zsh[2][rr][lane]);
        float ov = sigmoidf_(zsh[3][rr][lane]);
        int mm  = rr + 8 * half;
        int idx = mm * H_ + w * 16 + n;
        float cn = fv * c1[idx] + iv * gv;
        c1[idx] = cn;
        h1cur[idx] = f2bf(ov * tanhf(cn));
      }
    }
    grid_barrier(bar, nbar);
  }

  if (w < 4) {
#pragma unroll
    for (int r = 0; r < 8; ++r)
      decacc[((size_t)dw * 32 + lane) * 8 + r] = acc8[r];
  }
}

// deterministic final reduction to scalar
__global__ void finalize_kernel(float* out, const float* decacc, const int* lengths) {
  if (threadIdx.x != 0 || blockIdx.x != 0) return;
  float per[B_];
  for (int b = 0; b < B_; ++b) per[b] = 0.f;
  for (int dw = 0; dw < 16; ++dw)
    for (int lane = 0; lane < 32; ++lane)
      for (int r = 0; r < 8; ++r) {
        int mm = r + 8 * (lane >> 4);
        per[mm] += decacc[((size_t)dw * 32 + lane) * 8 + r];
      }
  float tot = 0.f;
  for (int b = 0; b < B_; ++b)
    tot += per[b] / ((float)(lengths[b] - 1) * (float)RANK_);
  out[0] = tot;
}

// ---------------------------------------------------------------------------
// launch
// ---------------------------------------------------------------------------
extern "C" void kernel_launch(void* const* d_in, const int* in_sizes, int n_in,
                              void* d_out, int out_size, void* d_ws, size_t ws_size,
                              hipStream_t stream) {
  const float* Ws     = (const float*)d_in[0];
  const int*   lens   = (const int*)d_in[1];
  const float* Wih0   = (const float*)d_in[2];
  const float* Whh0   = (const float*)d_in[3];
  const float* bih0   = (const float*)d_in[4];
  const float* bhh0   = (const float*)d_in[5];
  const float* Wih1   = (const float*)d_in[6];
  const float* Whh1   = (const float*)d_in[7];
  const float* bih1   = (const float*)d_in[8];
  const float* bhh1   = (const float*)d_in[9];
  const float* Wdec   = (const float*)d_in[10];
  const float* bdec   = (const float*)d_in[11];
  float* out = (float*)d_out;

  char* ws = (char*)d_ws;
  size_t off = 0;
  auto take = [&](size_t bytes) -> char* {
    char* p = ws + off;
    off = (off + bytes + 255) & ~(size_t)255;
    return p;
  };

  unsigned short* pw0  = (unsigned short*)take(256ull * KT0 * 512 * 2);  // 10.5 MB
  unsigned short* pw1  = (unsigned short*)take(256ull * KT1 * 512 * 2);  // 16.8 MB
  unsigned short* pwd  = (unsigned short*)take(16ull * KTD * 512 * 2);   // 0.5 MB
  unsigned short* xbf  = (unsigned short*)take((size_t)B_ * (T_ - 1) * RANK_ * 2);
  float*          b0   = (float*)take(G4_ * 4);
  float*          b1   = (float*)take(G4_ * 4);
  size_t state_off = off;                       // everything below is zeroed
  float*          c0   = (float*)take(B_ * H_ * 4);
  float*          c1   = (float*)take(B_ * H_ * 4);
  unsigned short* h0bf = (unsigned short*)take(2ull * B_ * H_ * 2);
  unsigned short* h1bf = (unsigned short*)take(2ull * B_ * H_ * 2);
  float*          dacc = (float*)take(16 * 32 * 8 * 4);
  unsigned int*   bar  = (unsigned int*)take(256);
  size_t state_bytes = off - state_off;

  // --- setup ---
  {
    long long tot0 = 256ll * KT0 * 512;
    pack_b_kernel<<<(unsigned)((tot0 + 255) / 256), 256, 0, stream>>>(
        pw0, Wih0, Whh0, RANK_, H_, KT0, tot0);
    long long tot1 = 256ll * KT1 * 512;
    pack_b_kernel<<<(unsigned)((tot1 + 255) / 256), 256, 0, stream>>>(
        pw1, Wih1, Whh1, H_, H_, KT1, tot1);
    long long totd = 16ll * KTD * 512;
    pack_b_kernel<<<(unsigned)((totd + 255) / 256), 256, 0, stream>>>(
        pwd, Wdec, Wdec, H_, H_, KTD, totd);
    long long totx = (long long)B_ * (T_ - 1) * RANK_;
    conv_x_kernel<<<(unsigned)((totx + 255) / 256), 256, 0, stream>>>(xbf, Ws);
    bias_kernel<<<(2 * G4_ + 255) / 256, 256, 0, stream>>>(b0, b1, bih0, bhh0, bih1, bhh1);
    long long zwords = (long long)(state_bytes / 4);
    zero_kernel<<<(unsigned)((zwords + 255) / 256), 256, 0, stream>>>(
        (unsigned int*)(ws + state_off), zwords);
  }

  // --- persistent fused LSTM (288 KB dynamic LDS weight cache per WGP) ---
  lstm_persistent<<<NWG, NTHR, LDS_BYTES, stream>>>(pw0, pw1, pwd, xbf, b0, b1,
                                                    c0, c1, h0bf, h1bf, dacc, bar,
                                                    Ws, lens, bdec);

  // --- scalar loss ---
  finalize_kernel<<<1, 64, 0, stream>>>(out, dacc, lens);
}
